// GlobalSpatialAttention_12051678233108
// MI455X (gfx1250) — matrix operand order repaired and checked by reading, not compiled, not run on hardware
//
#include <hip/hip_runtime.h>
#include <hip/hip_bf16.h>
#include <math.h>

// ---------------------------------------------------------------------------
// GlobalSpatialAttention for MI455X (gfx1250, wave32, WMMA)
// Flash-style two-pass global softmax: never materializes the 2304x2304 qk
// matrix (340MB/batch-set) -- recompute beats HBM at 23.3 TB/s vs WMMA rates.
// All GEMMs via v_wmma_f32_16x16x32_f16 (fp16 fragments, fp32 accumulate).
// v2: weights pre-converted to f16 (kills v_cvt in hot loops); k/v tiles
// staged in LDS once per block i-step (4 waves share them -> 4x less vmem).
// ---------------------------------------------------------------------------

#define NB   16
#define CIN  256
#define CR   64
#define HH   48
#define WW   48
#define HW   (HH * WW)      // 2304
#define JBLK 64             // output columns per 128-thread block (16 per wave)
#define NJB  (HW / JBLK)    // 36
#define KSP  72             // LDS row stride (halves) for 32x64 k tile (16B aligned, conflict-free)
#define VSP  40             // LDS row stride (halves) for 64x32 v tile (16B aligned)

typedef __attribute__((ext_vector_type(16))) _Float16 v16h;
typedef __attribute__((ext_vector_type(8)))  _Float16 v8h;
typedef __attribute__((ext_vector_type(8)))  float    v8f;

__device__ __forceinline__ v8f wmma32(v16h a, v16h b, v8f c) {
  // D = A(16x32,f16) * B(32x16,f16) + C(16x16,f32)
  return __builtin_amdgcn_wmma_f32_16x16x32_f16(
      /*neg_a=*/false, a, /*neg_b=*/false, b,
      /*c_mod=*/(short)0, c, /*reuse_a=*/false, /*reuse_b=*/false);
}

// A-fragment (16x32 f16). Source is row-major [M][K] with given stride.
// ISA layout: lanes 0-15 hold row M=l with K {0..7} then {16..23};
// lanes 16-31 hold row M=l with K {8..15} then {24..31}.
template <typename T>
__device__ __forceinline__ v16h load_a16(const T* base, int row0, int k0, int stride) {
  const int l  = threadIdx.x & 15;
  const int hl = (threadIdx.x >> 4) & 1;
  const T* p = base + (size_t)(row0 + l) * stride + k0 + hl * 8;
  v16h r;
#pragma unroll
  for (int i = 0; i < 8; ++i) {
    r[i]     = (_Float16)p[i];
    r[i + 8] = (_Float16)p[i + 16];
  }
  return r;
}

// B-fragment (32x16 f16). Source holds B transposed: element B[k][n] at
// base[n*stride + k].  ISA layout: lanes 0-15 = column n, K {0..7} then {16..23};
// lanes 16-31 = column n, K {8..15} then {24..31}.
template <typename T>
__device__ __forceinline__ v16h load_b16(const T* base, int col0, int k0, int stride) {
  const int l  = threadIdx.x & 15;
  const int hl = (threadIdx.x >> 4) & 1;
  const T* p = base + (size_t)(col0 + l) * stride + k0 + hl * 8;
  v16h r;
#pragma unroll
  for (int i = 0; i < 8; ++i) {
    r[i]     = (_Float16)p[i];
    r[i + 8] = (_Float16)p[i + 16];
  }
  return r;
}

// (m,s) online-softmax merge: s' = s1*e^(m1-m') + s2*e^(m2-m')
__device__ __forceinline__ void ms_merge(float& m, float& s, float om, float os) {
  float nm = fmaxf(m, om);
  s = s * __expf(m - nm) + os * __expf(om - nm);
  m = nm;
}

// ---------------------------------------------------------------------------
// Kernel 0: one-time fp32->f16 conversion of Wq|Wk|Wv|Watt (4 x 16384 elems).
// ---------------------------------------------------------------------------
__global__ __launch_bounds__(256) void k_cvtw(
    const float* __restrict__ Wq, const float* __restrict__ Wk,
    const float* __restrict__ Wv, const float* __restrict__ Watt,
    _Float16* __restrict__ w16) {
  int i = blockIdx.x * 256 + threadIdx.x;            // 0 .. 65535
  int seg = i >> 14, off = i & 16383;
  const float* src = (seg == 0) ? Wq : (seg == 1) ? Wk : (seg == 2) ? Wv : Watt;
  w16[i] = (_Float16)src[off];
}

// ---------------------------------------------------------------------------
// Kernel 1: q/k/v = W*x + b.  x tile is transposed+converted to f16 in LDS so
// B-fragments are contiguous ds reads; A-fragments are straight f16 b128 loads.
// q,k stored transposed [HW][CR] (packed 16B stores from the D-fragment),
// v stored [CR][HW].
// ---------------------------------------------------------------------------
__global__ __launch_bounds__(128) void k_qkv(
    const float* __restrict__ x,
    const _Float16* __restrict__ w16,                 // [Wq16|Wk16|Wv16|Watt16]
    const float* __restrict__ bq, const float* __restrict__ bk,
    const float* __restrict__ bv,
    _Float16* __restrict__ qT, _Float16* __restrict__ kT,
    _Float16* __restrict__ vv) {
  __shared__ _Float16 xs[JBLK][CIN + 8];   // +8 halves pad vs bank conflicts
  const int n  = blockIdx.y;
  const int j0 = blockIdx.x * JBLK;
  const float* xb = x + (size_t)n * CIN * HW;

  for (int idx = threadIdx.x; idx < CIN * JBLK; idx += 128) {
    int c = idx >> 6;            // / JBLK
    int j = idx & (JBLK - 1);
    xs[j][c] = (_Float16)xb[(size_t)c * HW + j0 + j];
  }
  __syncthreads();

  const int wave = threadIdx.x >> 5;
  const int jw   = wave * 16;               // wave's 16-col strip in block
  const int l    = threadIdx.x & 15;
  const int hl   = (threadIdx.x >> 4) & 1;
  const int j    = j0 + jw + l;             // this lane's global column

  const _Float16* Ws[3] = {w16, w16 + CR * CIN, w16 + 2 * CR * CIN};
  const float*    bs[3] = {bq, bk, bv};

#pragma unroll
  for (int r = 0; r < 3; ++r) {
    v8f acc[4];
#pragma unroll
    for (int t = 0; t < 4; ++t)
#pragma unroll
      for (int e = 0; e < 8; ++e) acc[t][e] = bs[r][t * 16 + hl * 8 + e];

    for (int k0 = 0; k0 < CIN; k0 += 32) {
      v16h bf = load_b16(&xs[0][0], jw, k0, CIN + 8);
#pragma unroll
      for (int t = 0; t < 4; ++t)
        acc[t] = wmma32(load_a16(Ws[r], t * 16, k0, CIN), bf, acc[t]);
    }

    if (r == 2) {  // v: [CR][HW]
#pragma unroll
      for (int t = 0; t < 4; ++t)
#pragma unroll
        for (int e = 0; e < 8; ++e) {
          int c = t * 16 + hl * 8 + e;
          vv[((size_t)n * CR + c) * HW + j] = (_Float16)acc[t][e];
        }
    } else {       // qT / kT: [HW][CR], 16B packed stores per tile
      _Float16* dst = (r == 0 ? qT : kT) + ((size_t)n * HW + j) * CR;
#pragma unroll
      for (int t = 0; t < 4; ++t) {
        v8h hp;
#pragma unroll
        for (int e = 0; e < 8; ++e) hp[e] = (_Float16)acc[t][e];
        *(v8h*)(dst + t * 16 + hl * 8) = hp;
      }
    }
  }
}

// Cooperative stage of a 32-row x 64-half kT tile into LDS (128 threads:
// 4 threads/row x 32B each, coalesced 128B rows).
__device__ __forceinline__ void stage_k(_Float16 (*ks)[KSP],
                                        const _Float16* kn, int i0) {
  const int r = threadIdx.x >> 2;
  const int p = (threadIdx.x & 3) * 16;
  const _Float16* src = kn + (size_t)(i0 + r) * CR + p;
  *(v8h*)&ks[r][p]     = *(const v8h*)src;
  *(v8h*)&ks[r][p + 8] = *(const v8h*)(src + 8);
}

// ---------------------------------------------------------------------------
// Kernel 2: streaming qk tiles -> per-block online (max, sumexp) partials.
// qk[i,j] = sum_c k[c,i] q[c,j]; one 16-wide j-strip per wave, i swept 0..HW.
// kT tile staged in LDS, shared by all 4 waves (4x less vmem traffic).
// ---------------------------------------------------------------------------
__global__ __launch_bounds__(128) void k_stats(
    const _Float16* __restrict__ qT, const _Float16* __restrict__ kT,
    float* __restrict__ partM, float* __restrict__ partS) {
  __shared__ _Float16 ks[32][KSP];
  __shared__ float sm[4], ss[4];
  const int n  = blockIdx.y;
  const int j0 = blockIdx.x * JBLK + (threadIdx.x >> 5) * 16;
  const _Float16* qn = qT + (size_t)n * HW * CR;
  const _Float16* kn = kT + (size_t)n * HW * CR;

  v16h bq0 = load_b16(qn, j0, 0, CR);
  v16h bq1 = load_b16(qn, j0, 32, CR);

  float m = -INFINITY, s = 0.f;
  for (int i0 = 0; i0 < HW; i0 += 32) {
    stage_k(ks, kn, i0);
    if (i0 + 32 < HW)
      __builtin_prefetch(kn + (size_t)(i0 + 32 + (threadIdx.x >> 2)) * CR, 0, 1);
    __syncthreads();

    v8f d0 = {}, d1 = {};
    d0 = wmma32(load_a16(&ks[0][0], 0, 0, KSP),   bq0, d0);
    d0 = wmma32(load_a16(&ks[0][0], 0, 32, KSP),  bq1, d0);
    d1 = wmma32(load_a16(&ks[0][0], 16, 0, KSP),  bq0, d1);
    d1 = wmma32(load_a16(&ks[0][0], 16, 32, KSP), bq1, d1);
    __syncthreads();

    float tm = d0[0];
#pragma unroll
    for (int e = 1; e < 8; ++e) tm = fmaxf(tm, d0[e]);
#pragma unroll
    for (int e = 0; e < 8; ++e) tm = fmaxf(tm, d1[e]);
    float nm  = fmaxf(m, tm);
    float acc = s * __expf(m - nm);
#pragma unroll
    for (int e = 0; e < 8; ++e)
      acc += __expf(d0[e] - nm) + __expf(d1[e] - nm);
    s = acc; m = nm;
  }

  // wave32 tree-merge of (m,s), then cross-wave via LDS
#pragma unroll
  for (int off = 16; off > 0; off >>= 1)
    ms_merge(m, s, __shfl_xor(m, off, 32), __shfl_xor(s, off, 32));

  if ((threadIdx.x & 31) == 0) { sm[threadIdx.x >> 5] = m; ss[threadIdx.x >> 5] = s; }
  __syncthreads();
  if (threadIdx.x == 0) {
    m = sm[0]; s = ss[0];
    for (int w = 1; w < 4; ++w) ms_merge(m, s, sm[w], ss[w]);
    partM[n * NJB + blockIdx.x] = m;
    partS[n * NJB + blockIdx.x] = s;
  }
}

// Kernel 3: fold NJB partials -> global (M, S) per batch (trivial).
__global__ void k_stats_final(const float* __restrict__ partM,
                              const float* __restrict__ partS,
                              float* __restrict__ Mg, float* __restrict__ Sg) {
  const int n = blockIdx.x;
  if (threadIdx.x == 0) {
    float m = -INFINITY, s = 0.f;
    for (int b = 0; b < NJB; ++b) ms_merge(m, s, partM[n * NJB + b], partS[n * NJB + b]);
    Mg[n] = m; Sg[n] = s;
  }
}

// ---------------------------------------------------------------------------
// Kernel 4: att^T = P^T * V^T, recomputing qk tiles.  The qk D-fragments map
// directly (per ISA 16-bit A layout) to the P^T A-fragment: element-wise exp,
// zero cross-lane traffic.  k and v tiles staged in LDS, shared by 4 waves.
// ---------------------------------------------------------------------------
__global__ __launch_bounds__(128) void k_attn(
    const _Float16* __restrict__ qT, const _Float16* __restrict__ kT,
    const _Float16* __restrict__ vv, const float* __restrict__ Mg,
    const float* __restrict__ Sg, _Float16* __restrict__ attT) {
  __shared__ _Float16 ks[32][KSP];
  __shared__ _Float16 vs[CR][VSP];
  const int n  = blockIdx.y;
  const int j0 = blockIdx.x * JBLK + (threadIdx.x >> 5) * 16;
  const _Float16* qn = qT + (size_t)n * HW * CR;
  const _Float16* kn = kT + (size_t)n * HW * CR;
  const _Float16* vn = vv + (size_t)n * CR * HW;
  const float M    = Mg[n];
  const float invS = 1.0f / Sg[n];

  v16h bq0 = load_b16(qn, j0, 0, CR);
  v16h bq1 = load_b16(qn, j0, 32, CR);

  v8f accT[4] = {};   // att^T tiles: rows j (16), cols c-tile t*16..t*16+15
  for (int i0 = 0; i0 < HW; i0 += 32) {
    stage_k(ks, kn, i0);
    {   // stage 64-row x 32-half v tile: 2 threads/row x 32B, coalesced
      const int r = threadIdx.x >> 1;
      const int p = (threadIdx.x & 1) * 16;
      const _Float16* src = vn + (size_t)r * HW + i0 + p;
      *(v8h*)&vs[r][p]     = *(const v8h*)src;
      *(v8h*)&vs[r][p + 8] = *(const v8h*)(src + 8);
    }
    __syncthreads();

    v8f d0 = {}, d1 = {};
    d0 = wmma32(load_a16(&ks[0][0], 0, 0, KSP),   bq0, d0);
    d0 = wmma32(load_a16(&ks[0][0], 0, 32, KSP),  bq1, d0);
    d1 = wmma32(load_a16(&ks[0][0], 16, 0, KSP),  bq0, d1);
    d1 = wmma32(load_a16(&ks[0][0], 16, 32, KSP), bq1, d1);

    v16h pa;  // P^T A-fragment (16x32): direct D->A remap, exp in fp32
#pragma unroll
    for (int e = 0; e < 8; ++e) {
      pa[e]     = (_Float16)__expf(d0[e] - M);
      pa[e + 8] = (_Float16)__expf(d1[e] - M);
    }
#pragma unroll
    for (int t = 0; t < 4; ++t)
      accT[t] = wmma32(pa, load_b16(&vs[0][0], t * 16, 0, VSP), accT[t]);
    __syncthreads();
  }

  const int l  = threadIdx.x & 15;
  const int hl = (threadIdx.x >> 4) & 1;
  _Float16* an = attT + (size_t)n * HW * CR;
#pragma unroll
  for (int t = 0; t < 4; ++t)
#pragma unroll
    for (int e = 0; e < 8; ++e)
      an[(size_t)(j0 + hl * 8 + e) * CR + t * 16 + l] =
          (_Float16)(accT[t][e] * invS);
}

// ---------------------------------------------------------------------------
// Kernel 5: out = g * (Watt*att + batt) + g.  [256x64]*[64x16] per wave-strip,
// B-fragments straight out of attT [HW][CR] layout, A from pre-converted f16.
// ---------------------------------------------------------------------------
__global__ __launch_bounds__(128) void k_out(
    const _Float16* __restrict__ attT, const _Float16* __restrict__ Watt16,
    const float* __restrict__ batt, const float* __restrict__ g,
    float* __restrict__ out) {
  const int n  = blockIdx.y;
  const int j0 = blockIdx.x * JBLK + (threadIdx.x >> 5) * 16;
  const _Float16* an = attT + (size_t)n * HW * CR;

  v16h b0 = load_b16(an, j0, 0, CR);
  v16h b1 = load_b16(an, j0, 32, CR);

  const int l  = threadIdx.x & 15;
  const int hl = (threadIdx.x >> 4) & 1;
  const int j  = j0 + l;

#pragma unroll 4
  for (int t = 0; t < 16; ++t) {    // 256 output channels
    v8f acc;
#pragma unroll
    for (int e = 0; e < 8; ++e) acc[e] = batt[t * 16 + hl * 8 + e];
    acc = wmma32(load_a16(Watt16, t * 16, 0, CR),  b0, acc);
    acc = wmma32(load_a16(Watt16, t * 16, 32, CR), b1, acc);
#pragma unroll
    for (int e = 0; e < 8; ++e) {
      int co = t * 16 + hl * 8 + e;
      size_t idx = ((size_t)n * CIN + co) * HW + j;
      float gv = g[idx];
      out[idx] = gv * acc[e] + gv;
    }
  }
}

// ---------------------------------------------------------------------------
extern "C" void kernel_launch(void* const* d_in, const int* in_sizes, int n_in,
                              void* d_out, int out_size, void* d_ws, size_t ws_size,
                              hipStream_t stream) {
  const float* x    = (const float*)d_in[0];  // feature_maps [N,CIN,H,W]
  const float* g    = (const float*)d_in[1];  // global_channel_output
  const float* Wq   = (const float*)d_in[2];
  const float* bq   = (const float*)d_in[3];
  const float* Wk   = (const float*)d_in[4];
  const float* bk   = (const float*)d_in[5];
  const float* Wv   = (const float*)d_in[6];
  const float* bv   = (const float*)d_in[7];
  const float* Watt = (const float*)d_in[8];
  const float* batt = (const float*)d_in[9];
  float* out = (float*)d_out;

  // Workspace carve-up (~19.1 MB of f16 + ~5 KB stats)
  const size_t halfs = (size_t)NB * HW * CR;   // 2,359,296 per tensor
  _Float16* qT   = (_Float16*)d_ws;
  _Float16* kT   = qT + halfs;
  _Float16* vv   = kT + halfs;
  _Float16* attT = vv + halfs;
  _Float16* w16  = attT + halfs;               // [Wq16|Wk16|Wv16|Watt16]
  float* partM = (float*)(w16 + 4 * CR * CIN);
  float* partS = partM + (size_t)NB * NJB;
  float* Mg    = partS + (size_t)NB * NJB;
  float* Sg    = Mg + NB;

  dim3 grid(NJB, NB), blk(128);
  k_cvtw<<<dim3(256), dim3(256), 0, stream>>>(Wq, Wk, Wv, Watt, w16);
  k_qkv<<<grid, blk, 0, stream>>>(x, w16, bq, bk, bv, qT, kT, vv);
  k_stats<<<grid, blk, 0, stream>>>(qT, kT, partM, partS);
  k_stats_final<<<dim3(NB), dim3(32), 0, stream>>>(partM, partS, Mg, Sg);
  k_attn<<<grid, blk, 0, stream>>>(qT, kT, vv, Mg, Sg, attT);
  k_out<<<grid, blk, 0, stream>>>(attT, w16 + 3 * CR * CIN, batt, g, out);
}
